// PairSelfAttention_77232101916715
// MI455X (gfx1250) — matrix operand order repaired and checked by reading, not compiled, not run on hardware
//
#include <hip/hip_runtime.h>
#include <hip/hip_bf16.h>

typedef __attribute__((ext_vector_type(16))) _Float16 v16h;
typedef __attribute__((ext_vector_type(8)))  float    v8f;

#define BB 8
#define HH 8
#define SS 1024
#define DD 64
#define QT 128   // q rows per block
#define KT 32    // keys per tile
#define NW 8     // waves per block
#define KROW 136 // padded K' row length in halves (128 + 8)
#define VROW 40  // padded V^T row length in halves (32 + 8)

// ---------------- Kernel 1: IoU-argmax partner (replicates buggy reference IoU) ----------
__global__ __launch_bounds__(256) void pairs_kernel(const float* __restrict__ centers,
                                                    int* __restrict__ partner) {
    int wave = threadIdx.x >> 5, lane = threadIdx.x & 31;
    int row = blockIdx.x * 8 + wave;           // global (b,q) row
    int b = row >> 10, q = row & (SS - 1);
    const float* ci = centers + (size_t)(b * SS + q) * 4;
    float cx = ci[0], cy = ci[1], hh = ci[2], ww = ci[3];
    float ax0 = cx - 0.5f * ww, ay0 = cy - 0.5f * hh;
    float ax1 = cx + 0.5f * ww, ay1 = cy + 0.5f * hh;
    float aarea = (ax1 - ax0) * (ay1 - ay0);

    float best = -3.4e38f; int bidx = 0;
    for (int t = 0; t < SS / 32; ++t) {
        int j = t * 32 + lane;
        const float* cj = centers + (size_t)(b * SS + j) * 4;
        float jcx = cj[0], jcy = cj[1], jh = cj[2], jw = cj[3];
        float bx0 = jcx - 0.5f * jw, by0 = jcy - 0.5f * jh;
        float bx1 = jcx + 0.5f * jw, by1 = jcy + 0.5f * jh;
        float ix0 = fmaxf(ax0, bx0), iy0 = fmaxf(ay0, by0);
        float ix1 = fmaxf(ax1, bx1), iy1 = fmaxf(ay1, by1);   // reference uses MAX here (bug kept)
        float iw = fmaxf(ix1 - ix0, 0.f), ih = fmaxf(iy1 - iy0, 0.f);
        float inter = iw * ih;
        float barea = (bx1 - bx0) * (by1 - by0);
        float uni = aarea + barea - inter;
        float iou = inter / uni - ((j == q) ? 1.0f : 0.0f);
        if (iou > best) { best = iou; bidx = j; }              // ascending j -> first max per lane
    }
    #pragma unroll
    for (int m = 16; m >= 1; m >>= 1) {
        float ov = __shfl_xor(best, m, 32);
        int   oi = __shfl_xor(bidx, m, 32);
        if (ov > best || (ov == best && oi < bidx)) { best = ov; bidx = oi; }
    }
    if (lane == 0) partner[row] = bidx;
}

// ---------------- Kernel 2: order pair by L1 (= w + h) ----------------------------------
__global__ __launch_bounds__(256) void p01_kernel(const float* __restrict__ centers,
                                                  const int* __restrict__ partner,
                                                  int* __restrict__ p0, int* __restrict__ p1) {
    int i = blockIdx.x * 256 + threadIdx.x;            // 0 .. B*S-1
    int b = i >> 10, q = i & (SS - 1);
    int j = partner[i];
    const float* ci = centers + (size_t)i * 4;
    const float* cj = centers + (size_t)(b * SS + j) * 4;
    float l1i = ci[2] + ci[3];                          // h + w (both positive)
    float l1j = cj[2] + cj[3];
    bool keep = (l1i >= l1j);
    p0[i] = keep ? q : j;
    p1[i] = keep ? j : q;
}

// ---------------- Kernel 3: transposed flash attention, gathered 128-dim Q'/K' ----------
__device__ __forceinline__ void load_tile(const float* __restrict__ Kb,
                                          const float* __restrict__ Vb,
                                          const int* __restrict__ p0b,
                                          const int* __restrict__ p1b,
                                          int kb, int skey, int sseg, int koff, int vd0,
                                          float (&kreg)[16], float (&vreg)[8]) {
    int rp = (sseg < 4) ? p0b[kb + skey] : p1b[kb + skey];
    const float* srck = Kb + (size_t)rp * DD + koff;
    #pragma unroll
    for (int i = 0; i < 16; ++i) kreg[i] = srck[i];
    const float* srcv = Vb + (size_t)(kb + skey) * DD + vd0;
    #pragma unroll
    for (int i = 0; i < 8; ++i) vreg[i] = srcv[i];
}

__device__ __forceinline__ void store_tile(_Float16* __restrict__ lk,
                                           _Float16* __restrict__ lv,
                                           int skey, int sseg, int koff, int vd0,
                                           const float (&kreg)[16], const float (&vreg)[8]) {
    _Float16* dk = lk + skey * KROW + ((sseg < 4) ? 0 : 64) + koff;
    #pragma unroll
    for (int i = 0; i < 16; ++i) dk[i] = (_Float16)kreg[i];
    #pragma unroll
    for (int i = 0; i < 8; ++i) lv[(vd0 + i) * VROW + skey] = (_Float16)vreg[i];
}

__global__ __launch_bounds__(256) void attn_kernel(const float* __restrict__ Q,
                                                   const float* __restrict__ K,
                                                   const float* __restrict__ V,
                                                   const int* __restrict__ p0,
                                                   const int* __restrict__ p1,
                                                   float* __restrict__ out) {
    __shared__ _Float16 lk[2][KT * KROW];   // gathered K' tile: 32 keys x 128 (f16, padded)
    __shared__ _Float16 lv[2][DD * VROW];   // V^T tile: 64 d x 32 keys (f16, padded)

    int bid = blockIdx.x;
    int qt = bid & 7;                 // q tile (S/QT == 8)
    int h  = (bid >> 3) & 7;
    int b  = bid >> 6;
    int tid = threadIdx.x;
    int wave = tid >> 5, lane = tid & 31;
    int lm = lane & 15;
    bool lo = (lane < 16);
    int ko    = lo ? 0 : 16;   // A-fragment contiguous K half
    int kb16  = lo ? 0 : 16;   // B-fragment contiguous K half
    int rbase = lo ? 0 : 8;    // C-fragment row base

    const float* Qb = Q + (size_t)(b * HH + h) * SS * DD;
    const float* Kb = K + (size_t)(b * HH + h) * SS * DD;
    const float* Vb = V + (size_t)(b * HH + h) * SS * DD;
    const int* p0b = p0 + b * SS;
    const int* p1b = p1 + b * SS;

    // ---- Q'^T as B operand: per lane column q = qrow, contiguous k from global
    int qrow = qt * QT + wave * 16 + lm;
    int r0 = p0b[qrow], r1 = p1b[qrow];
    v16h qb[4];
    #pragma unroll
    for (int c = 0; c < 4; ++c) {
        const float* src = Qb + (size_t)((c < 2) ? r0 : r1) * DD + (c & 1) * 32 + kb16;
        #pragma unroll
        for (int i = 0; i < 16; ++i) qb[c][i] = (_Float16)src[i];
    }

    // staging assignment
    int skey = tid >> 3, sseg = tid & 7;
    int koff = (sseg & 3) * 16;
    int vd0  = sseg * 8;

    float m = -1e30f, l = 0.0f;
    v8f o[4];
    #pragma unroll
    for (int t = 0; t < 4; ++t)
        #pragma unroll
        for (int e = 0; e < 8; ++e) o[t][e] = 0.0f;

    const float scale = 0.08838834764831845f;   // 1/sqrt(2*D) = 1/sqrt(128)

    float kreg[16], vreg[8];
    load_tile(Kb, Vb, p0b, p1b, 0, skey, sseg, koff, vd0, kreg, vreg);
    store_tile(lk[0], lv[0], skey, sseg, koff, vd0, kreg, vreg);
    __syncthreads();

    for (int kt = 0; kt < SS / KT; ++kt) {
        int bufi = kt & 1;
        bool more = (kt + 1 < SS / KT);
        if (more)   // issue next tile's global loads early; latency hidden by compute
            load_tile(Kb, Vb, p0b, p1b, (kt + 1) * KT, skey, sseg, koff, vd0, kreg, vreg);

        const _Float16* ck = lk[bufi];
        const _Float16* cv = lv[bufi];

        // ---- all 8 K' A-fragments up front, then an uninterrupted 8-WMMA chain
        v16h ak[8];
        #pragma unroll
        for (int g = 0; g < 2; ++g)
            #pragma unroll
            for (int c = 0; c < 4; ++c) {
                const _Float16* p = ck + (g * 16 + lm) * KROW + c * 32 + ko;
                #pragma unroll
                for (int i = 0; i < 8; ++i) { ak[g*4+c][i] = p[i]; ak[g*4+c][8+i] = p[8+i]; }
            }
        v8f s0, s1;
        #pragma unroll
        for (int e = 0; e < 8; ++e) { s0[e] = 0.0f; s1[e] = 0.0f; }
        #pragma unroll
        for (int c = 0; c < 4; ++c) {
            s0 = __builtin_amdgcn_wmma_f32_16x16x32_f16(false, ak[c],     false, qb[c],
                                                        (short)0, s0, false, false);
            s1 = __builtin_amdgcn_wmma_f32_16x16x32_f16(false, ak[4 + c], false, qb[c],
                                                        (short)0, s1, false, false);
        }

        // ---- per-lane online softmax: this lane owns query column q = lm,
        //      holding keys {rbase..rbase+7} U {16+rbase..16+rbase+7}; partner lane^16 the rest
        float sc[16];
        #pragma unroll
        for (int j = 0; j < 8; ++j) { sc[j] = s0[j] * scale; sc[8 + j] = s1[j] * scale; }
        float mx = sc[0];
        #pragma unroll
        for (int i = 1; i < 16; ++i) mx = fmaxf(mx, sc[i]);
        mx = fmaxf(mx, __shfl_xor(mx, 16, 32));
        float mn = fmaxf(m, mx);
        float alpha = __expf(m - mn);
        float e[16], rs = 0.0f;
        #pragma unroll
        for (int i = 0; i < 16; ++i) { e[i] = __expf(sc[i] - mn); rs += e[i]; }
        rs += __shfl_xor(rs, 16, 32);
        l = l * alpha + rs;
        m = mn;
        #pragma unroll
        for (int t = 0; t < 4; ++t)
            #pragma unroll
            for (int j = 0; j < 8; ++j) o[t][j] *= alpha;

        // ---- assemble P~ B-fragment (32 keys x 16 q) directly from registers
        v16h pb;
        #pragma unroll
        for (int j = 0; j < 8; ++j) {
            float send = lo ? e[8 + j] : e[j];
            float recv = __shfl_xor(send, 16, 32);
            pb[j]     = (_Float16)(lo ? e[j] : recv);     // keys kb16 + j
            pb[8 + j] = (_Float16)(lo ? recv : e[8 + j]); // keys kb16 + 8 + j
        }

        // ---- O^T += V^T * P~   (A = V^T d-tiles from LDS, B = P~ from registers)
        #pragma unroll
        for (int t = 0; t < 4; ++t) {
            v16h av;
            const _Float16* p = cv + (t * 16 + lm) * VROW + ko;
            #pragma unroll
            for (int i = 0; i < 8; ++i) { av[i] = p[i]; av[8 + i] = p[8 + i]; }
            o[t] = __builtin_amdgcn_wmma_f32_16x16x32_f16(false, av, false, pb,
                                                          (short)0, o[t], false, false);
        }

        if (more)
            store_tile(lk[bufi ^ 1], lv[bufi ^ 1], skey, sseg, koff, vd0, kreg, vreg);
        __syncthreads();
    }

    // ---- epilogue: O^T[d, q]; lane writes 4 runs of 8 consecutive d for its query
    float inv = 1.0f / l;
    float* ob = out + ((size_t)(b * SS + qrow) * HH + h) * DD;
    #pragma unroll
    for (int t = 0; t < 4; ++t)
        #pragma unroll
        for (int j = 0; j < 8; ++j)
            ob[t * 16 + rbase + j] = o[t][j] * inv;
}

extern "C" void kernel_launch(void* const* d_in, const int* in_sizes, int n_in,
                              void* d_out, int out_size, void* d_ws, size_t ws_size,
                              hipStream_t stream) {
    const float* q = (const float*)d_in[0];
    const float* k = (const float*)d_in[1];
    const float* v = (const float*)d_in[2];
    const float* c = (const float*)d_in[3];

    int* partner = (int*)d_ws;            // B*S ints
    int* p0 = partner + BB * SS;          // B*S ints
    int* p1 = p0 + BB * SS;               // B*S ints

    pairs_kernel<<<BB * SS / 8, 256, 0, stream>>>(c, partner);
    p01_kernel<<<BB * SS / 256, 256, 0, stream>>>(c, partner, p0, p1);
    attn_kernel<<<BB * HH * (SS / QT), 256, 0, stream>>>(q, k, v, p0, p1, (float*)d_out);
}
